// SpMiddlePillarEncoderVgg_33629593928125
// MI455X (gfx1250) — matrix operand order repaired and checked by reading, not compiled, not run on hardware
//
#include <hip/hip_runtime.h>
#include <hip/hip_bf16.h>

typedef __attribute__((ext_vector_type(16))) _Float16 v16h;
typedef __attribute__((ext_vector_type(8)))  _Float16 v8h;
typedef __attribute__((ext_vector_type(8)))  float    v8f;

#define NYX   512
#define BATCH 2

// ---------------------------------------------------------------------------
// Scatter pillar features (fp32 atomics) into dense NHWC grid + build mask m1
// ---------------------------------------------------------------------------
__global__ void scatter_kernel(const float* __restrict__ feats,
                               const int* __restrict__ coords,
                               float* __restrict__ dense,
                               _Float16* __restrict__ m1, int nvox) {
  int v = blockIdx.x;
  int c = threadIdx.x;            // 64 channels, one per thread
  if (v >= nvox) return;
  int b = coords[4 * v + 0];
  int y = coords[4 * v + 2];
  int x = coords[4 * v + 3];
  size_t p = (((size_t)b * NYX + y) * NYX + x);
  atomicAdd(&dense[p * 64 + c], feats[(size_t)v * 64 + c]);
  if (c == 0) m1[p] = (_Float16)1.0f;
}

__global__ void cvt_f32_to_f16(const float* __restrict__ in,
                               _Float16* __restrict__ out, size_t n) {
  size_t i = (size_t)blockIdx.x * blockDim.x + threadIdx.x;
  if (i < n) out[i] = (_Float16)in[i];
}

// OIHW fp32 -> [Cout][tap*Cin + ci] f16 (tap = ky*3+kx), matching WMMA A tiling
__global__ void pack_weights(const float* __restrict__ w,
                             _Float16* __restrict__ out, int cin, int cout) {
  int n = cout * 9 * cin;
  int i = blockIdx.x * blockDim.x + threadIdx.x;
  if (i >= n) return;
  int k9  = 9 * cin;
  int co  = i / k9;
  int r   = i % k9;
  int tap = r / cin;
  int ci  = r % cin;
  out[i] = (_Float16)w[((size_t)co * cin + ci) * 9 + tap];
}

// 3x3 stride-2 "any nonzero" pooling of the occupancy mask
__global__ void pool_mask_kernel(const _Float16* __restrict__ in,
                                 _Float16* __restrict__ out, int Hi, int Wi) {
  int Ho = Hi >> 1, Wo = Wi >> 1;
  int i = blockIdx.x * blockDim.x + threadIdx.x;
  int tot = BATCH * Ho * Wo;
  if (i >= tot) return;
  int xo = i % Wo; int t = i / Wo;
  int yo = t % Ho; int b = t / Ho;
  float s = 0.f;
  for (int dy = 0; dy < 3; ++dy)
    for (int dx = 0; dx < 3; ++dx) {
      int yi = 2 * yo + dy - 1, xi = 2 * xo + dx - 1;
      if (yi >= 0 && yi < Hi && xi >= 0 && xi < Wi)
        s += (float)in[((size_t)b * Hi + yi) * Wi + xi];
    }
  out[((size_t)b * Ho + yo) * Wo + xo] = (_Float16)(s > 0.f ? 1.f : 0.f);
}

// ---------------------------------------------------------------------------
// Implicit-GEMM 3x3 conv + BN (+ReLU) + mask, f16 in / f16 out (+opt fp32 NCHW)
// Block = 8 waves = 256 threads; tile = 32 output pixels x COUT channels.
// Wave w: pixel-half (w&1), cout tile-set (w>>1). Double-buffered LDS stage,
// one barrier per 32-wide K chunk. All tile control flow is wave-uniform
// (readfirstlane) so EXEC is all-ones at every v_wmma.
// ---------------------------------------------------------------------------
template <int CIN, int COUT, int STRIDE, bool RELU>
__global__ __launch_bounds__(256, 2)
void conv3x3_bn_wmma(const _Float16* __restrict__ in,    // [B][Hi][Wi][CIN] f16
                     const _Float16* __restrict__ wgt,   // [COUT][9*CIN]    f16
                     const float* __restrict__ scale,
                     const float* __restrict__ bias,
                     const _Float16* __restrict__ mask,  // [B][Ho][Wo] or null
                     _Float16* __restrict__ out,         // [B][Ho][Wo][COUT]
                     float* __restrict__ out32,          // [B][COUT][Ho][Wo] or null
                     int Hi, int Wi) {
  constexpr int K9     = 9 * CIN;
  constexpr int NCHUNK = K9 / 32;        // K chunks of 32 (CIN % 32 == 0)
  constexpr int CPT    = CIN / 32;       // chunks per 3x3 tap
  constexpr int NTILES = COUT / 16;
  constexpr int TPW    = (NTILES + 3) / 4;

  const int Ho = Hi / STRIDE, Wo = Wi / STRIDE;
  const int lane  = threadIdx.x & 31;
  const int wave  = __builtin_amdgcn_readfirstlane(threadIdx.x >> 5); // uniform
  const int phalf = wave & 1;            // which 16-pixel half of the 32-tile
  const int tw    = wave >> 1;           // cout tile-set 0..3
  const int n     = lane & 15;           // pixel column within 16-tile
  const int hi16  = lane >> 4;           // 0 or 1 (lane half)
  const int xo0   = blockIdx.x * 32;
  const int yo    = blockIdx.y;
  const int b     = blockIdx.z;
  const int xo    = xo0 + phalf * 16 + n;

  // [buffer][pixel-half][pixel][k]
  __shared__ __attribute__((aligned(64))) _Float16 lds[2][2][16][32];

  v8f acc[TPW] = {};

  // staging role: wave w loads k-rows 4w..4w+3 for all 32 pixels (8B each)
  const int sn  = lane;                  // pixel 0..31
  const int sg  = wave;                  // k group 0..7
  const int sxo = xo0 + sn;
  const _Float16* inb = in + (size_t)b * Hi * Wi * CIN;

  auto stage_load = [&](int chunk) -> unsigned long long {
    const int tap   = chunk / CPT;
    const int cbase = (chunk % CPT) * 32;
    const int yi = yo * STRIDE + tap / 3 - 1;
    const int xi = sxo * STRIDE + tap % 3 - 1;
    unsigned long long v = 0ull;
    if (yi >= 0 && yi < Hi && xi >= 0 && xi < Wi)
      v = *(const unsigned long long*)(inb + ((size_t)yi * Wi + xi) * CIN +
                                       cbase + sg * 4);
    return v;
  };
  auto stage_store = [&](int nb, unsigned long long v) {
    *(unsigned long long*)(&lds[nb][sn >> 4][sn & 15][sg * 4]) = v;
  };

  stage_store(0, stage_load(0));
  __syncthreads();

#pragma unroll 1
  for (int chunk = 0; chunk < NCHUNK; ++chunk) {
    const int cur = chunk & 1;
    // issue next chunk's global load early (latency hidden behind WMMAs)
    const bool have_next = (chunk + 1) < NCHUNK;
    unsigned long long nv = 0ull;
    if (have_next) nv = stage_load(chunk + 1);
    // prefetch two chunks ahead (global_prefetch_b8)
    if (chunk + 2 < NCHUNK) {
      const int tap2 = (chunk + 2) / CPT, cb2 = ((chunk + 2) % CPT) * 32;
      const int yi2 = yo * STRIDE + tap2 / 3 - 1;
      const int xi2 = sxo * STRIDE + tap2 % 3 - 1;
      if (yi2 >= 0 && yi2 < Hi && xi2 >= 0 && xi2 < Wi)
        __builtin_prefetch(inb + ((size_t)yi2 * Wi + xi2) * CIN + cb2 + sg * 4, 0, 0);
    }

    // B fragment: lanes 0-15 hold K 0..15, lanes 16-31 hold K 16..31, col = n
    const v16h bfrag = *(const v16h*)(&lds[cur][phalf][n][hi16 * 16]);
    const int kbase = chunk * 32;
    const int koff  = hi16 * 16;         // A layout: hi half holds K+16 / K+24
#pragma unroll
    for (int t = 0; t < TPW; ++t) {
      const int tile = tw + t * 4;
      if (NTILES < 4 && tile >= NTILES) continue;   // uniform (scalar branch)
      const _Float16* wrow = wgt + (size_t)(tile * 16 + n) * K9 + kbase;
      v8h alo = *(const v8h*)(wrow + koff);
      v8h ahi = *(const v8h*)(wrow + 8 + koff);
      v16h afrag;
#pragma unroll
      for (int i = 0; i < 8; ++i) { afrag[i] = alo[i]; afrag[i + 8] = ahi[i]; }
      acc[t] = __builtin_amdgcn_wmma_f32_16x16x32_f16(
          false, afrag, false, bfrag, (short)0, acc[t], false, false);
    }

    if (have_next) stage_store(cur ^ 1, nv);
    __syncthreads();
  }

  // Epilogue: BN + (ReLU) + mask; D layout: VGPR r -> cout = tile*16+hi16*8+r
  const float mf = mask ? (float)mask[((size_t)b * Ho + yo) * Wo + xo] : 1.0f;
  _Float16* obase = out + (((size_t)b * Ho + yo) * Wo + xo) * COUT;
#pragma unroll
  for (int t = 0; t < TPW; ++t) {
    const int tile = tw + t * 4;
    if (NTILES < 4 && tile >= NTILES) continue;
    const int co0 = tile * 16 + hi16 * 8;
    const v8f sc8 = *(const v8f*)(scale + co0);     // contiguous b128 pair
    const v8f bi8 = *(const v8f*)(bias + co0);
    v8h hv;
    v8f vf;
#pragma unroll
    for (int r = 0; r < 8; ++r) {
      float v = acc[t][r] * sc8[r] + bi8[r];
      if (RELU) v = fmaxf(v, 0.f);
      v *= mf;
      vf[r] = v;
      hv[r] = (_Float16)v;
    }
    *(v8h*)(obase + co0) = hv;                      // 16B contiguous f16 store
    if (out32) {                                    // hoisted uniform branch
      float* o32 = out32 + (((size_t)b * COUT + co0) * Ho + yo) * Wo + xo;
      const size_t cs = (size_t)Ho * Wo;
#pragma unroll
      for (int r = 0; r < 8; ++r) o32[r * cs] = vf[r];
    }
  }
}

// ---------------------------------------------------------------------------
extern "C" void kernel_launch(void* const* d_in, const int* in_sizes, int n_in,
                              void* d_out, int out_size, void* d_ws, size_t ws_size,
                              hipStream_t stream) {
  const float* feats  = (const float*)d_in[0];
  const int*   coords = (const int*)d_in[1];
  const float *W[10], *S[10], *Bv[10];
  for (int j = 0; j < 10; ++j) {
    W[j]  = (const float*)d_in[2 + 3 * j];
    S[j]  = (const float*)d_in[3 + 3 * j];
    Bv[j] = (const float*)d_in[4 + 3 * j];
  }
  const int nvox = in_sizes[0] / 64;

  static const int CINS[10]  = {64, 32, 32, 64, 64, 128, 128, 256, 256, 256};
  static const int COUTS[10] = {32, 32, 64, 64, 128, 128, 256, 256, 256, 256};

  char* ws = (char*)d_ws;
  size_t off = 0;
  float*    dense = (float*)(ws + off);    off += (size_t)BATCH * NYX * NYX * 64 * 4;
  _Float16* bufA  = (_Float16*)(ws + off); off += (size_t)BATCH * NYX * NYX * 64 * 2;
  _Float16* bufB  = (_Float16*)(ws + off); off += (size_t)BATCH * NYX * NYX * 64 * 2;
  _Float16* m1    = (_Float16*)(ws + off); off += (size_t)BATCH * 512 * 512 * 2;
  _Float16* m2    = (_Float16*)(ws + off); off += (size_t)BATCH * 256 * 256 * 2;
  _Float16* m3    = (_Float16*)(ws + off); off += (size_t)BATCH * 128 * 128 * 2;
  _Float16* m4    = (_Float16*)(ws + off); off += (size_t)BATCH * 64 * 64 * 2;
  _Float16* wp[10];
  for (int j = 0; j < 10; ++j) {
    wp[j] = (_Float16*)(ws + off);
    off += (size_t)COUTS[j] * 9 * CINS[j] * 2;
  }

  hipMemsetAsync(dense, 0, (size_t)BATCH * NYX * NYX * 64 * 4, stream);
  hipMemsetAsync(m1, 0, (size_t)BATCH * 512 * 512 * 2, stream);

  scatter_kernel<<<nvox, 64, 0, stream>>>(feats, coords, dense, m1, nvox);

  { size_t n = (size_t)BATCH * NYX * NYX * 64;
    cvt_f32_to_f16<<<(unsigned)((n + 255) / 256), 256, 0, stream>>>(dense, bufA, n); }

  for (int j = 0; j < 10; ++j) {
    int n = COUTS[j] * 9 * CINS[j];
    pack_weights<<<(n + 255) / 256, 256, 0, stream>>>(W[j], wp[j], CINS[j], COUTS[j]);
  }

  pool_mask_kernel<<<(BATCH * 256 * 256 + 255) / 256, 256, 0, stream>>>(m1, m2, 512, 512);
  pool_mask_kernel<<<(BATCH * 128 * 128 + 255) / 256, 256, 0, stream>>>(m2, m3, 256, 256);
  pool_mask_kernel<<<(BATCH * 64 * 64 + 255) / 256, 256, 0, stream>>>(m3, m4, 128, 128);

  float* x3_out = (float*)d_out;
  float* x4_out = x3_out + (size_t)BATCH * 128 * 128 * 128;
  float* x5_out = x4_out + (size_t)BATCH * 256 * 64 * 64;

#define CONV(CI, CO, ST, RL, IN, OUT, J, MSK, O32, HI, WI)                          \
  conv3x3_bn_wmma<CI, CO, ST, RL>                                                   \
      <<<dim3((WI / ST) / 32, (HI / ST), BATCH), 256, 0, stream>>>(                 \
          IN, wp[J], S[J], Bv[J], MSK, OUT, O32, HI, WI)

  CONV(64, 32, 1, false, bufA, bufB, 0, m1, (float*)nullptr, 512, 512);             // 1a
  CONV(32, 32, 1, true,  bufB, bufA, 1, m1, (float*)nullptr, 512, 512);             // 1b
  CONV(32, 64, 2, true,  bufA, bufB, 2, m2, (float*)nullptr, 512, 512);             // 2a
  CONV(64, 64, 1, true,  bufB, bufA, 3, m2, (float*)nullptr, 256, 256);             // 2b
  CONV(64, 128, 2, true, bufA, bufB, 4, m3, (float*)nullptr, 256, 256);             // 3a
  CONV(128, 128, 1, true, bufB, bufA, 5, m3, x3_out, 128, 128);                     // 3b -> x3
  CONV(128, 256, 2, true, bufA, bufB, 6, m4, (float*)nullptr, 128, 128);            // 4a
  CONV(256, 256, 1, true, bufB, bufA, 7, m4, x4_out, 64, 64);                       // 4b -> x4
  CONV(256, 256, 2, true, bufA, bufB, 8, (const _Float16*)nullptr,
       (float*)nullptr, 64, 64);                                                    // 5a
  CONV(256, 256, 1, true, bufB, bufA, 9, (const _Float16*)nullptr,
       x5_out, 32, 32);                                                             // 5b -> x5
#undef CONV
}
